// MLA_1563368095776
// MI455X (gfx1250) — matrix operand order repaired and checked by reading, not compiled, not run on hardware
//
#include <hip/hip_runtime.h>
#include <hip/hip_bf16.h>

// ---------------------------------------------------------------------------
// MLA forward (DeepSeek-style with lightning indexer + top-k sparse attention)
// gfx1250 / MI455X: wave32, bf16 WMMA (v_wmma_f32_16x16x32_bf16), fp32 accum.
// bf16 operand tensors are materialized once so every WMMA fragment is a
// contiguous aligned 128-bit load (no per-element conversion on hot paths).
// ---------------------------------------------------------------------------

typedef __attribute__((ext_vector_type(16))) __bf16 v16bf;
typedef __attribute__((ext_vector_type(8)))  __bf16 v8bf;
typedef __attribute__((ext_vector_type(8)))  float  v8f;

#define TT   2048   // T (sequence length), also KV
#define HNUM 16
#define ROPE 64
#define QKD  192    // NOPE + ROPE
#define VD   128
#define IHN  4
#define TOPK 512

__device__ __forceinline__ unsigned short bfbits(float f) {
  unsigned u = __float_as_uint(f);
  u += 0x7FFFu + ((u >> 16) & 1u);          // round-to-nearest-even
  return (unsigned short)(u >> 16);
}
__device__ __forceinline__ __bf16 tobf(float f) {
  unsigned short hs = bfbits(f);
  return __builtin_bit_cast(__bf16, hs);
}
__device__ __forceinline__ unsigned pk2(float lo, float hi) {
  return (unsigned)bfbits(lo) | ((unsigned)bfbits(hi) << 16);
}

__device__ __forceinline__ v8f wmma_bf16(v16bf a, v16bf b, v8f c) {
  return __builtin_amdgcn_wmma_f32_16x16x32_bf16(
      false, a, false, b, (short)0, c, false, false);
}

__device__ __forceinline__ v16bf cat16(v8bf lo, v8bf hi) {
  return __builtin_shufflevector(lo, hi, 0, 1, 2, 3, 4, 5, 6, 7,
                                         8, 9, 10, 11, 12, 13, 14, 15);
}

__device__ __forceinline__ unsigned f2key(float f) {
  unsigned u = __float_as_uint(f);
  return (u & 0x80000000u) ? ~u : (u | 0x80000000u);   // order-preserving
}

// ---------------------------------------------------------------------------
// Generic GEMM: C[M,N] = A[M,K] @ W[N,K]^T + bias.  64x128 tile, 4 waves,
// each wave 32x64 (2 A-frags x 4 B-frags -> 8 WMMA per K-step).
// fp32 -> packed bf16 pairs in LDS; fragments read as ds_load_b128.
// M and K must be multiples of 64/32; N is bounds-checked.
// ---------------------------------------------------------------------------
__global__ __launch_bounds__(128) void gemm_bias(
    const float* __restrict__ A, int lda,
    const float* __restrict__ W,            // [N,K] row-major
    const float* __restrict__ bias,
    float* __restrict__ C, int ldc,
    int M, int N, int K)
{
  __shared__ __align__(64) unsigned As32[64][16];   // row m, packed k-pairs
  __shared__ __align__(64) unsigned Ws32[128][16];  // row n, packed k-pairs
  const int m0 = blockIdx.y * 64, n0 = blockIdx.x * 128;
  const int tid = threadIdx.x;
  const int lane = tid & 31, wave = tid >> 5;
  const int wm = (wave >> 1) * 32, wn = (wave & 1) * 64;
  const int half = lane >> 4, l15 = lane & 15;

  v8f acc[2][4] = {};
  for (int k0 = 0; k0 < K; k0 += 32) {
    if (k0 + 32 < K) {   // L2-resident operands: near-cache prefetch next tile
      __builtin_prefetch(&A[(size_t)(m0 + (tid >> 1)) * lda + k0 + 32], 0, 3);
      if (n0 + tid < N)
        __builtin_prefetch(&W[(size_t)(n0 + tid) * K + k0 + 32], 0, 3);
    }
    for (int i = tid; i < 64 * 16; i += 128) {
      int r = i >> 4, c2 = (i & 15) * 2;
      const float* p = &A[(size_t)(m0 + r) * lda + k0 + c2];
      As32[r][c2 >> 1] = pk2(p[0], p[1]);
    }
    for (int i = tid; i < 128 * 16; i += 128) {
      int n = i >> 4, c2 = (i & 15) * 2;
      float v0 = 0.f, v1 = 0.f;
      if (n0 + n < N) {
        const float* p = &W[(size_t)(n0 + n) * K + k0 + c2];
        v0 = p[0]; v1 = p[1];
      }
      Ws32[n][c2 >> 1] = pk2(v0, v1);
    }
    __syncthreads();
    v16bf afr[2], bfr[4];
    #pragma unroll
    for (int t = 0; t < 2; ++t) {
      const v8bf* ar = (const v8bf*)&As32[wm + t * 16 + l15][0];
      afr[t] = cat16(ar[2 * half], ar[2 * half + 1]);
    }
    #pragma unroll
    for (int b = 0; b < 4; ++b) {
      const v16bf* br = (const v16bf*)&Ws32[wn + b * 16 + l15][0];
      bfr[b] = br[half];
    }
    #pragma unroll
    for (int a = 0; a < 2; ++a)
      #pragma unroll
      for (int b = 0; b < 4; ++b)
        acc[a][b] = wmma_bf16(afr[a], bfr[b], acc[a][b]);
    __syncthreads();
  }
  #pragma unroll
  for (int a = 0; a < 2; ++a)
    #pragma unroll
    for (int b = 0; b < 4; ++b) {
      int n = n0 + wn + b * 16 + l15;
      if (n >= N) continue;
      float bv = bias[n];
      #pragma unroll
      for (int r = 0; r < 8; ++r) {
        int m = m0 + wm + a * 16 + half * 8 + r;
        C[(size_t)m * ldc + n] = acc[a][b][r] + bv;
      }
    }
}

// ---------------------------------------------------------------------------
// In-place LayerNorm over last dim D (one block per row).
// ---------------------------------------------------------------------------
__global__ __launch_bounds__(128) void ln_rows(
    float* __restrict__ x, const float* __restrict__ g,
    const float* __restrict__ b, int D, int ld)
{
  float* p = x + (size_t)blockIdx.x * ld;
  const int tid = threadIdx.x;
  float s = 0.f, s2 = 0.f;
  for (int i = tid; i < D; i += 128) { float v = p[i]; s += v; s2 += v * v; }
  __shared__ float r1[128], r2[128];
  r1[tid] = s; r2[tid] = s2;
  __syncthreads();
  for (int o = 64; o > 0; o >>= 1) {
    if (tid < o) { r1[tid] += r1[tid + o]; r2[tid] += r2[tid + o]; }
    __syncthreads();
  }
  float mean = r1[0] / (float)D;
  float var  = r2[0] / (float)D - mean * mean;
  float rs   = rsqrtf(var + 1e-5f);
  for (int i = tid; i < D; i += 128) p[i] = (p[i] - mean) * rs * g[i] + b[i];
}

// Per-128-chunk LN of kv_heads rows (first 16 chunks = k_norm, last 16 = v_norm)
__global__ __launch_bounds__(128) void ln_kvh(
    float* __restrict__ kvh,
    const float* __restrict__ kg, const float* __restrict__ kb2,
    const float* __restrict__ vg, const float* __restrict__ vb2)
{
  const int c = blockIdx.x, row = c >> 5, j = c & 31;
  float* p = kvh + (size_t)row * 4096 + (size_t)j * 128;
  const float* g = (j < 16) ? kg : vg;
  const float* b = (j < 16) ? kb2 : vb2;
  const int tid = threadIdx.x;
  float v = p[tid];
  __shared__ float r1[128], r2[128];
  r1[tid] = v; r2[tid] = v * v;
  __syncthreads();
  for (int o = 64; o > 0; o >>= 1) {
    if (tid < o) { r1[tid] += r1[tid + o]; r2[tid] += r2[tid + o]; }
    __syncthreads();
  }
  float mean = r1[0] * (1.f / 128.f);
  float var  = r2[0] * (1.f / 128.f) - mean * mean;
  float rs   = rsqrtf(var + 1e-5f);
  p[tid] = (v - mean) * rs * g[tid] + b[tid];
}

// ---------------------------------------------------------------------------
// Gather + rotary kernels building bf16 Q[h][t][192], K[h][s][192],
// V^T[h][d][s], honoring the reference's flat-reshape head mapping.
// ---------------------------------------------------------------------------
__global__ __launch_bounds__(128) void build_q(
    const float* __restrict__ qln, __bf16* __restrict__ Qb)
{
  const int blk = blockIdx.x, h = blk >> 11, t = blk & (TT - 1);
  __bf16* dst = Qb + (size_t)blk * QKD;
  const int tid = threadIdx.x;
  {
    size_t i = ((size_t)h * TT + t) * 128 + tid;
    int row = (int)(i >> 11), col = (int)(i & 2047);
    dst[tid] = tobf(qln[(size_t)row * 3072 + col]);
  }
  if (tid < 32) {
    const int j = tid;
    size_t i2 = ((size_t)h * TT + t) * 64 + 2 * j;
    int row = (int)(i2 >> 10), col = 2048 + (int)(i2 & 1023);
    float x1 = qln[(size_t)row * 3072 + col];
    float x2 = qln[(size_t)row * 3072 + col + 1];
    float fr = __powf(10000.f, -(float)(2 * j) / 64.f);
    float a = (float)t * fr, c = __cosf(a), s = __sinf(a);
    dst[128 + 2 * j]     = tobf(x1 * c - x2 * s);
    dst[128 + 2 * j + 1] = tobf(x1 * s + x2 * c);
  }
}

__global__ __launch_bounds__(128) void build_k(
    const float* __restrict__ kvh, const float* __restrict__ z,
    __bf16* __restrict__ Kb)
{
  const int blk = blockIdx.x, h = blk >> 11, s = blk & (TT - 1);
  __bf16* dst = Kb + (size_t)blk * QKD;
  const int tid = threadIdx.x;
  {
    size_t i = ((size_t)h * TT + s) * 128 + tid;
    int row = (int)(i >> 11), col = (int)(i & 2047);
    dst[tid] = tobf(kvh[(size_t)row * 4096 + col]);
  }
  if (tid < 32) {
    const int j = tid;
    float x1 = z[(size_t)s * 576 + 512 + 2 * j];
    float x2 = z[(size_t)s * 576 + 512 + 2 * j + 1];
    float fr = __powf(10000.f, -(float)(2 * j) / 64.f);
    float a = (float)s * fr, c = __cosf(a), sn = __sinf(a);
    dst[128 + 2 * j]     = tobf(x1 * c - x2 * sn);
    dst[128 + 2 * j + 1] = tobf(x1 * sn + x2 * c);
  }
}

__global__ __launch_bounds__(128) void build_vt(
    const float* __restrict__ kvh, __bf16* __restrict__ Vt)
{
  const int blk = blockIdx.x, h = blk >> 11, s = blk & (TT - 1);
  const int d = threadIdx.x;
  size_t i = ((size_t)h * TT + s) * 128 + d;
  int row = (int)(i >> 11), col = 2048 + (int)(i & 2047);
  Vt[((size_t)h * 128 + d) * TT + s] = tobf(kvh[(size_t)row * 4096 + col]);
}

// Indexer q/k: split + rotary -> bf16 iqf (T,4,128), ikf (T,128)
__global__ __launch_bounds__(128) void build_indexer(
    const float* __restrict__ iq, const float* __restrict__ ik,
    __bf16* __restrict__ iqf, __bf16* __restrict__ ikf)
{
  const int t = blockIdx.x, tid = threadIdx.x;
  for (int i = tid; i < 256; i += 128) {
    int h = i >> 6, d = i & 63;
    iqf[(size_t)t * 512 + h * 128 + d] = tobf(iq[(size_t)t * 512 + h * 64 + d]);
  }
  {
    int h = tid >> 5, j = tid & 31;
    float x1 = iq[(size_t)t * 512 + 256 + h * 64 + 2 * j];
    float x2 = iq[(size_t)t * 512 + 256 + h * 64 + 2 * j + 1];
    float fr = __powf(10000.f, -(float)(2 * j) / 64.f);
    float a = (float)t * fr, c = __cosf(a), s = __sinf(a);
    iqf[(size_t)t * 512 + h * 128 + 64 + 2 * j]     = tobf(x1 * c - x2 * s);
    iqf[(size_t)t * 512 + h * 128 + 64 + 2 * j + 1] = tobf(x1 * s + x2 * c);
  }
  if (tid < 64) ikf[(size_t)t * 128 + tid] = tobf(ik[(size_t)t * 128 + tid]);
  if (tid < 32) {
    int j = tid;
    float x1 = ik[(size_t)t * 128 + 64 + 2 * j];
    float x2 = ik[(size_t)t * 128 + 64 + 2 * j + 1];
    float fr = __powf(10000.f, -(float)(2 * j) / 64.f);
    float a = (float)t * fr, c = __cosf(a), s = __sinf(a);
    ikf[(size_t)t * 128 + 64 + 2 * j]     = tobf(x1 * c - x2 * s);
    ikf[(size_t)t * 128 + 64 + 2 * j + 1] = tobf(x1 * s + x2 * c);
  }
}

// ---------------------------------------------------------------------------
// Indexer scores: S[t][s] = sum_h w[t][h]*relu(q_h . k), causal-masked.
// One wave per 16x16 tile; fragments are direct 128-bit vector loads.
// ---------------------------------------------------------------------------
__global__ __launch_bounds__(32) void idx_scores(
    const __bf16* __restrict__ iqf, const __bf16* __restrict__ ikf,
    const float* __restrict__ iw, float* __restrict__ S)
{
  const int t0 = blockIdx.y * 16, s0 = blockIdx.x * 16;
  const int lane = threadIdx.x, half = lane >> 4, l15 = lane & 15;
  if (s0 > t0 + 15) {
    #pragma unroll
    for (int r = 0; r < 8; ++r)
      S[(size_t)(t0 + half * 8 + r) * TT + s0 + l15] = -1e9f;
    return;
  }
  float out[8] = {};
  #pragma unroll
  for (int h = 0; h < IHN; ++h) {
    v8f acc = {};
    #pragma unroll
    for (int kk = 0; kk < 128; kk += 32) {
      const v8bf* ap = (const v8bf*)(iqf + (size_t)(t0 + l15) * 512 + h * 128 + kk);
      v16bf af = cat16(ap[2 * half], ap[2 * half + 1]);
      const v16bf* bp = (const v16bf*)(ikf + (size_t)(s0 + l15) * 128 + kk);
      acc = wmma_bf16(af, bp[half], acc);
    }
    #pragma unroll
    for (int r = 0; r < 8; ++r) {
      int t = t0 + half * 8 + r;
      float v = acc[r];
      out[r] += iw[t * 4 + h] * (v > 0.f ? v : 0.f);
    }
  }
  #pragma unroll
  for (int r = 0; r < 8; ++r) {
    int t = t0 + half * 8 + r, s = s0 + l15;
    S[(size_t)t * TT + s] = (s <= t) ? out[r] : -1e9f;
  }
}

// ---------------------------------------------------------------------------
// Top-k (512) per row via 4-pass 8-bit radix select -> bit-packed mask
// (64 words per row, causal folded into the bits).
// ---------------------------------------------------------------------------
__global__ __launch_bounds__(256) void topk_mask(
    const float* __restrict__ S, unsigned* __restrict__ Mw)
{
  const int t = blockIdx.x;
  const float* row = S + (size_t)t * TT;
  const int valid = t + 1;
  unsigned thresh = 0;   // 0 => everything valid passes
  if (valid > TOPK) {
    __shared__ unsigned hist[256];
    __shared__ unsigned sh_prefix;
    __shared__ int sh_k;
    unsigned prefix = 0;
    int kk = TOPK;
    for (int level = 0; level < 4; ++level) {
      const int shift = 24 - level * 8;
      hist[threadIdx.x] = 0;
      __syncthreads();
      for (int s = threadIdx.x; s < valid; s += 256) {
        unsigned key = f2key(row[s]);
        bool match = (level == 0) || ((key >> (shift + 8)) == (prefix >> (shift + 8)));
        if (match) atomicAdd(&hist[(key >> shift) & 255u], 1u);
      }
      __syncthreads();
      if (threadIdx.x == 0) {
        int cum = 0, bsel = 255;
        for (; bsel >= 0; --bsel) {
          int c = (int)hist[bsel];
          if (cum + c >= kk) break;
          cum += c;
        }
        if (bsel < 0) bsel = 0;
        sh_prefix = prefix | ((unsigned)bsel << shift);
        sh_k = kk - cum;
      }
      __syncthreads();
      prefix = sh_prefix; kk = sh_k;
      __syncthreads();
    }
    thresh = prefix;
  }
  for (int w = threadIdx.x; w < TT / 32; w += 256) {
    unsigned bits = 0;
    #pragma unroll 4
    for (int b = 0; b < 32; ++b) {
      int s = w * 32 + b;
      bool on = (s <= t) && (f2key(row[s]) >= thresh);
      bits |= (unsigned)on << b;
    }
    Mw[(size_t)t * (TT / 32) + w] = bits;
  }
}

// ---------------------------------------------------------------------------
// Flash attention: one wave per (head, 16-row tile). QK^T and P*V via WMMA,
// online softmax with shfl_xor row reductions, P restaged through LDS (bf16).
// All fragments are contiguous 128-bit loads (K row-major, V transposed);
// sparse mask read as one b32 word per row per 32-column step.
// ---------------------------------------------------------------------------
__global__ __launch_bounds__(32) void attn_fwd(
    const __bf16* __restrict__ Qb, const __bf16* __restrict__ Kb,
    const __bf16* __restrict__ Vt, const unsigned* __restrict__ Mw,
    float* __restrict__ Ob)
{
  const int h = blockIdx.y, t0 = blockIdx.x * 16;
  const int lane = threadIdx.x, half = lane >> 4, l15 = lane & 15;
  const float scale = 0.072168784f;   // 1/sqrt(192)

  // Q fragments (A-layout), K-dim 192 -> 6 chains of 32
  v16bf qf[6];
  const __bf16* Qh = Qb + ((size_t)h * TT + t0) * QKD;
  #pragma unroll
  for (int kk = 0; kk < 6; ++kk) {
    const v8bf* qp = (const v8bf*)(Qh + (size_t)l15 * QKD + kk * 32);
    qf[kk] = cat16(qp[2 * half], qp[2 * half + 1]);
  }

  v8f acc_o[8] = {};
  float rmax[8], rsum[8];
  #pragma unroll
  for (int r = 0; r < 8; ++r) { rmax[r] = -1e30f; rsum[r] = 0.f; }

  __shared__ __align__(64) __bf16 Pl[16][32];
  const __bf16* Kh  = Kb + (size_t)h * TT * QKD;
  const __bf16* Vth = Vt + (size_t)h * 128 * TT;
  const int tmax = t0 + 15;

  for (int s0 = 0; s0 <= tmax; s0 += 32) {
    // S tile 16x32 = two 16x16 WMMA chains
    v8f s_acc[2];
    #pragma unroll
    for (int j = 0; j < 2; ++j) {
      v8f a = {};
      const __bf16* kb = Kh + (size_t)(s0 + j * 16 + l15) * QKD;
      #pragma unroll
      for (int kk = 0; kk < 6; ++kk) {
        const v16bf* bp = (const v16bf*)(kb + kk * 32);
        a = wmma_bf16(qf[kk], bp[half], a);
      }
      s_acc[j] = a;
    }
    // masked online softmax update (per row r of this lane's half)
    float p0[8], p1[8];
    #pragma unroll
    for (int r = 0; r < 8; ++r) {
      const int t = t0 + half * 8 + r;
      const unsigned mw = Mw[(size_t)t * (TT / 32) + (s0 >> 5)];
      float vA = ((mw >> l15) & 1u)        ? s_acc[0][r] * scale : -1e30f;
      float vB = ((mw >> (16 + l15)) & 1u) ? s_acc[1][r] * scale : -1e30f;
      float mx = fmaxf(vA, vB);
      #pragma unroll
      for (int o = 1; o < 16; o <<= 1) mx = fmaxf(mx, __shfl_xor(mx, o, 32));
      float nm = fmaxf(rmax[r], mx);
      float f  = __expf(rmax[r] - nm);
      float ea = (vA > -1e29f) ? __expf(vA - nm) : 0.f;
      float eb = (vB > -1e29f) ? __expf(vB - nm) : 0.f;
      float ss = ea + eb;
      #pragma unroll
      for (int o = 1; o < 16; o <<= 1) ss += __shfl_xor(ss, o, 32);
      rsum[r] = rsum[r] * f + ss;
      rmax[r] = nm;
      p0[r] = ea; p1[r] = eb;
      #pragma unroll
      for (int j = 0; j < 8; ++j) acc_o[j][r] *= f;
    }
    // restage P (C-layout) -> LDS -> A-layout fragment (b128 reads)
    #pragma unroll
    for (int r = 0; r < 8; ++r) {
      Pl[half * 8 + r][l15]      = tobf(p0[r]);
      Pl[half * 8 + r][16 + l15] = tobf(p1[r]);
    }
    __syncthreads();
    const v8bf* pp = (const v8bf*)&Pl[l15][0];
    v16bf pf = cat16(pp[2 * half], pp[2 * half + 1]);
    __syncthreads();
    // O += P @ V   (8 N-tiles of 16; V^T rows give contiguous B-fragments)
    #pragma unroll
    for (int j = 0; j < 8; ++j) {
      const v16bf* vp = (const v16bf*)(Vth + (size_t)(j * 16 + l15) * TT + s0);
      acc_o[j] = wmma_bf16(pf, vp[half], acc_o[j]);
    }
  }
  // write out: Ob[t][h*128 + j*16 + n]
  #pragma unroll
  for (int j = 0; j < 8; ++j)
    #pragma unroll
    for (int r = 0; r < 8; ++r) {
      const int t = t0 + half * 8 + r;
      float d = (rsum[r] > 1e-20f) ? rsum[r] : 1e-20f;
      Ob[(size_t)t * 2048 + h * 128 + j * 16 + l15] = acc_o[j][r] / d;
    }
}

// ---------------------------------------------------------------------------
extern "C" void kernel_launch(void* const* d_in, const int* in_sizes, int n_in,
                              void* d_out, int out_size, void* d_ws, size_t ws_size,
                              hipStream_t stream) {
  (void)in_sizes; (void)n_in; (void)out_size; (void)ws_size;
  const float* x       = (const float*)d_in[0];
  const float* Wq_a_w  = (const float*)d_in[1];
  const float* Wq_a_b  = (const float*)d_in[2];
  const float* Wq_b_w  = (const float*)d_in[3];
  const float* Wq_b_b  = (const float*)d_in[4];
  const float* Wkv_a_w = (const float*)d_in[5];
  const float* Wkv_a_b = (const float*)d_in[6];
  const float* Wkv_b_w = (const float*)d_in[7];
  const float* Wkv_b_b = (const float*)d_in[8];
  const float* W_o_w   = (const float*)d_in[9];
  const float* W_o_b   = (const float*)d_in[10];
  const float* q_g     = (const float*)d_in[11];
  const float* q_b     = (const float*)d_in[12];
  const float* kv_g    = (const float*)d_in[13];
  const float* kv_b    = (const float*)d_in[14];
  const float* k_g     = (const float*)d_in[15];
  const float* k_b     = (const float*)d_in[16];
  const float* v_g     = (const float*)d_in[17];
  const float* v_b     = (const float*)d_in[18];
  const float* iWq_w   = (const float*)d_in[19];
  const float* iWq_b   = (const float*)d_in[20];
  const float* iWk_w   = (const float*)d_in[21];
  const float* iWk_b   = (const float*)d_in[22];
  const float* iWw_w   = (const float*)d_in[23];
  const float* iWw_b   = (const float*)d_in[24];
  const float* iqn_g   = (const float*)d_in[25];
  const float* iqn_b   = (const float*)d_in[26];
  const float* ikn_g   = (const float*)d_in[27];
  const float* ikn_b   = (const float*)d_in[28];
  const float* iwn_g   = (const float*)d_in[29];
  const float* iwn_b   = (const float*)d_in[30];
  float* out = (float*)d_out;

  char* base = (char*)d_ws;
  size_t off = 0;
  auto alloc = [&](size_t bytes) -> void* {
    void* p = base + off;
    off = (off + bytes + 255) & ~(size_t)255;
    return p;
  };
  float*  zbuf = (float*)alloc((size_t)TT * 576 * 4);
  float*  qa   = (float*)alloc((size_t)TT * 1536 * 4);
  float*  qbuf = (float*)alloc((size_t)TT * 3072 * 4);
  float*  kvh  = (float*)alloc((size_t)TT * 4096 * 4);
  __bf16* Qb   = (__bf16*)alloc((size_t)HNUM * TT * QKD * 2);
  __bf16* Kbuf = (__bf16*)alloc((size_t)HNUM * TT * QKD * 2);
  __bf16* Vt   = (__bf16*)alloc((size_t)HNUM * 128 * TT * 2);
  float*  iq   = (float*)alloc((size_t)TT * 512 * 4);
  __bf16* iqf  = (__bf16*)alloc((size_t)TT * 512 * 2);
  float*  ikb  = (float*)alloc((size_t)TT * 128 * 4);
  __bf16* ikf  = (__bf16*)alloc((size_t)TT * 128 * 2);
  float*  iwb  = (float*)alloc((size_t)TT * 4 * 4);
  float*  S    = (float*)alloc((size_t)TT * TT * 4);
  float*  Ob   = (float*)alloc((size_t)TT * 2048 * 4);
  unsigned* Mw = (unsigned*)alloc((size_t)TT * (TT / 32) * 4);

  const dim3 blk(128);
  // kv_a projection + LN  (z: T x 576)
  gemm_bias<<<dim3(5, 32), blk, 0, stream>>>(x, 2048, Wkv_a_w, Wkv_a_b, zbuf, 576, TT, 576, 2048);
  ln_rows<<<TT, blk, 0, stream>>>(zbuf, kv_g, kv_b, 576, 576);
  // q_a -> q_b + LN  (q: T x 3072)
  gemm_bias<<<dim3(12, 32), blk, 0, stream>>>(x, 2048, Wq_a_w, Wq_a_b, qa, 1536, TT, 1536, 2048);
  gemm_bias<<<dim3(24, 32), blk, 0, stream>>>(qa, 1536, Wq_b_w, Wq_b_b, qbuf, 3072, TT, 3072, 1536);
  ln_rows<<<TT, blk, 0, stream>>>(qbuf, q_g, q_b, 3072, 3072);
  // kv_b projection (kv = z[:, :512]) + per-head LN
  gemm_bias<<<dim3(32, 32), blk, 0, stream>>>(zbuf, 576, Wkv_b_w, Wkv_b_b, kvh, 4096, TT, 4096, 512);
  ln_kvh<<<TT * 32, blk, 0, stream>>>(kvh, k_g, k_b, v_g, v_b);
  // build bf16 Q / K / V^T (gather + rotary)
  build_q<<<HNUM * TT, blk, 0, stream>>>(qbuf, Qb);
  build_k<<<HNUM * TT, blk, 0, stream>>>(kvh, zbuf, Kbuf);
  build_vt<<<HNUM * TT, blk, 0, stream>>>(kvh, Vt);
  // indexer projections + LN
  gemm_bias<<<dim3(4, 32), blk, 0, stream>>>(x, 2048, iWq_w, iWq_b, iq, 512, TT, 512, 2048);
  ln_rows<<<TT, blk, 0, stream>>>(iq, iqn_g, iqn_b, 512, 512);
  gemm_bias<<<dim3(1, 32), blk, 0, stream>>>(zbuf, 576, iWk_w, iWk_b, ikb, 128, TT, 128, 512);
  ln_rows<<<TT, blk, 0, stream>>>(ikb, ikn_g, ikn_b, 128, 128);
  gemm_bias<<<dim3(1, 32), blk, 0, stream>>>(x, 2048, iWw_w, iWw_b, iwb, 4, TT, 4, 2048);
  ln_rows<<<TT, blk, 0, stream>>>(iwb, iwn_g, iwn_b, 4, 4);
  build_indexer<<<TT, blk, 0, stream>>>(iq, ikb, iqf, ikf);
  // indexer scores + top-k bitmask
  idx_scores<<<dim3(TT / 16, TT / 16), 32, 0, stream>>>(iqf, ikf, iwb, S);
  topk_mask<<<TT, 256, 0, stream>>>(S, Mw);
  // sparse flash attention
  attn_fwd<<<dim3(TT / 16, HNUM), 32, 0, stream>>>(Qb, Kbuf, Vt, Mw, Ob);
  // output projection
  gemm_bias<<<dim3(16, 32), blk, 0, stream>>>(Ob, 2048, W_o_w, W_o_b, out, 2048, TT, 2048, 2048);
}
